// DGCRN_51110110822949
// MI455X (gfx1250) — matrix operand order repaired and checked by reading, not compiled, not run on hardware
//
#include <hip/hip_runtime.h>
#include <hip/hip_bf16.h>
#include <math.h>

// Problem constants (from reference)
#define B_    32
#define T_    12
#define N_    1024
#define DIN_  2
#define HID_  64
#define EMB_  8
#define HOR_  12
#define CIN_  66            // DIN+HID == DOUT+DY+HID (real columns per batch)
#define CINP_ 80            // padded to 5 WMMA column tiles
#define LDP_  (B_*CINP_)    // 2560 padded columns, layout [N][LDP] (f32) / [LDP][N] (bf16)
#define GATE_ (2*HID_)      // 128
#define NT_   5             // column tiles per wave (16*5 = 80 = CINP_)

typedef __bf16 bf16_t;
typedef __attribute__((ext_vector_type(16))) __bf16 v16bf;
typedef __attribute__((ext_vector_type(8)))  __bf16 v8bf;
typedef __attribute__((ext_vector_type(2)))  __bf16 v2bf;
typedef __attribute__((ext_vector_type(8)))  float  v8f;

// ---------------------------------------------------------------- utilities
__global__ void zero_kernel(float* __restrict__ p, int n) {
    int i = blockIdx.x * blockDim.x + threadIdx.x;
    if (i < n) p[i] = 0.0f;
}

// ------------------------------------------------------- feature pack/concat
// Writes Xf[m][j] (row-major, f32) and XT[j][m] (transposed, bf16), j = b*CINP + c,
// with zeros in pad columns (c >= CIN_). Thread order: m fastest -> XT coalesced.
// mode 0 (encoder): c<2 -> x[b,t,m,c]            ; else h (optionally * z)
// mode 1 (decoder): c==0 -> go[b,m]; c==1 -> ycov; else h (optionally * z)
__global__ void pack_kernel(int mode, int t,
                            const float* __restrict__ x,
                            const float* __restrict__ ycov,
                            const float* __restrict__ go,
                            const float* __restrict__ h,
                            const float* __restrict__ zr,   // non-null => multiply h by z
                            float* __restrict__ Xf, bf16_t* __restrict__ XT) {
    int idx = blockIdx.x * blockDim.x + threadIdx.x;
    if (idx >= N_ * LDP_) return;
    int m = idx & (N_ - 1);
    int j = idx >> 10;            // N_ == 1024
    int b = j / CINP_, c = j % CINP_;
    float v = 0.0f;
    if (c < CIN_) {
        if (mode == 0) {
            if (c < DIN_) {
                v = x[(((size_t)b * T_ + t) * N_ + m) * DIN_ + c];
            } else {
                float hv = h[((size_t)b * N_ + m) * HID_ + (c - DIN_)];
                if (zr) hv *= zr[((size_t)b * N_ + m) * GATE_ + (c - DIN_)];
                v = hv;
            }
        } else {
            if (c == 0) {
                v = go[(size_t)b * N_ + m];
            } else if (c == 1) {
                v = ycov[((size_t)b * HOR_ + t) * N_ + m];
            } else {
                float hv = h[((size_t)b * N_ + m) * HID_ + (c - 2)];
                if (zr) hv *= zr[((size_t)b * N_ + m) * GATE_ + (c - 2)];
                v = hv;
            }
        }
    }
    Xf[(size_t)m * LDP_ + j] = v;
    XT[(size_t)j * N_ + m]   = (bf16_t)v;
}

// ------------------------------------------------- WMMA graph-support GEMM
// Y[n][col] = sum_m S[bz][n][m] * X[m][col].
// One wave computes a 16x(16*NT_) strip; K = N_ = 1024, 32 per wmma.
// A fragment (16x32 bf16): lane holds K = 16h..16h+15 of its row (same order as B)
//   -> two contiguous 16B runs of the row-major S row.
// B fragment (32x16 bf16): lane holds K = 16h..16h+15 for its column
//   -> one contiguous 32B run of the K-major (transposed) X column.
__global__ void __launch_bounds__(32)
sgemm_wmma(const bf16_t* __restrict__ S, size_t sStride,
           const bf16_t* __restrict__ XT,   // [LDP][N] transposed bf16
           float* __restrict__ Yf,          // [N][LDP] row-major f32
           bf16_t* __restrict__ YT,         // [LDP][N] transposed bf16 (or null)
           int batchColStride) {
    const int lane = threadIdx.x;            // full wave, EXEC all-1s
    const int half = lane >> 4;
    const int lan  = lane & 15;
    const int rowBase = blockIdx.y * 16;
    const int bz   = blockIdx.z;
    const int col0 = bz * batchColStride + blockIdx.x * (16 * NT_);

    const bf16_t* srow = S + (size_t)bz * sStride + (size_t)(rowBase + lan) * N_;
    const bf16_t* bcol[NT_];
#pragma unroll
    for (int tt = 0; tt < NT_; ++tt)
        bcol[tt] = XT + (size_t)(col0 + 16 * tt + lan) * N_ + 16 * half;

    v8f acc[NT_];
#pragma unroll
    for (int tt = 0; tt < NT_; ++tt) acc[tt] = (v8f){};

    for (int kk = 0; kk < N_; kk += 32) {
        const bf16_t* ap = srow + kk + 16 * half;
        v8bf alo = *(const v8bf*)(ap);          // K: 16h .. 16h+7
        v8bf ahi = *(const v8bf*)(ap + 8);      // K: 16h+8 .. 16h+15
        v16bf a = __builtin_shufflevector(alo, ahi,
                    0, 1, 2, 3, 4, 5, 6, 7, 8, 9, 10, 11, 12, 13, 14, 15);
#pragma unroll
        for (int tt = 0; tt < NT_; ++tt) {
            const bf16_t* bp = bcol[tt] + kk;
            v8bf blo = *(const v8bf*)(bp);      // K: 16h .. 16h+7
            v8bf bhi = *(const v8bf*)(bp + 8);  // K: 16h+8 .. 16h+15
            v16bf b = __builtin_shufflevector(blo, bhi,
                        0, 1, 2, 3, 4, 5, 6, 7, 8, 9, 10, 11, 12, 13, 14, 15);
            acc[tt] = __builtin_amdgcn_wmma_f32_16x16x32_bf16(false, a, false, b,
                                                              (short)0, acc[tt], false, false);
        }
    }

    // D layout: row = v + 8*half, col = lane%16
#pragma unroll
    for (int tt = 0; tt < NT_; ++tt) {
        int col = col0 + 16 * tt + lan;
#pragma unroll
        for (int v = 0; v < 8; ++v)
            Yf[(size_t)(rowBase + v + 8 * half) * LDP_ + col] = acc[tt][v];
        if (YT) {
            bf16_t* yp = YT + (size_t)col * N_ + rowBase + 8 * half;
#pragma unroll
            for (int v = 0; v < 8; v += 2) {
                v2bf pk = { (bf16_t)acc[tt][v], (bf16_t)acc[tt][v + 1] };
                *(v2bf*)(yp + v) = pk;          // rows consecutive -> 4B store
            }
        }
    }
}

// -------------------------------------------------------- gate path (z, r)
// zr[b,n,o] = sigmoid( sum_c  X*Wg[c] + T1*Wg[66+c] + (2*T2 - X)*Wg[132+c] + bg )
__global__ void gates_kernel(const float* __restrict__ Xf, const float* __restrict__ T1,
                             const float* __restrict__ T2,
                             const float* __restrict__ Wg, const float* __restrict__ bg,
                             float* __restrict__ zr) {
    int idx = blockIdx.x * blockDim.x + threadIdx.x;
    if (idx >= B_ * N_ * GATE_) return;
    int o  = idx % GATE_;
    int bn = idx / GATE_;
    int b = bn / N_, n = bn % N_;
    int jb = b * CINP_;
    float acc = bg[o];
    for (int c = 0; c < CIN_; ++c) {
        size_t off = (size_t)n * LDP_ + jb + c;
        float xv = Xf[off], t1 = T1[off], t2 = T2[off];
        acc += xv * Wg[c * GATE_ + o]
             + t1 * Wg[(CIN_ + c) * GATE_ + o]
             + (2.0f * t2 - xv) * Wg[(2 * CIN_ + c) * GATE_ + o];
    }
    zr[idx] = 1.0f / (1.0f + expf(-acc));
}

// --------------------------------------------- candidate path + GRU update
__global__ void cand_kernel(const float* __restrict__ Xf, const float* __restrict__ U1,
                            const float* __restrict__ U2,
                            const float* __restrict__ Wu, const float* __restrict__ bu,
                            const float* __restrict__ zr, float* __restrict__ h) {
    int idx = blockIdx.x * blockDim.x + threadIdx.x;
    if (idx >= B_ * N_ * HID_) return;
    int o  = idx % HID_;
    int bn = idx / HID_;
    int b = bn / N_, n = bn % N_;
    int jb = b * CINP_;
    float acc = bu[o];
    for (int c = 0; c < CIN_; ++c) {
        size_t off = (size_t)n * LDP_ + jb + c;
        float xv = Xf[off], u1 = U1[off], u2 = U2[off];
        acc += xv * Wu[c * HID_ + o]
             + u1 * Wu[(CIN_ + c) * HID_ + o]
             + (2.0f * u2 - xv) * Wu[(2 * CIN_ + c) * HID_ + o];
    }
    float hc = tanhf(acc);
    float r  = zr[(size_t)bn * GATE_ + HID_ + o];
    float ho = h[idx];
    h[idx] = r * ho + (1.0f - r) * hc;
}

// --------------------------------------------------------- hyper embedding
__global__ void ne_kernel(const float* __restrict__ h, const float* __restrict__ hyp_W,
                          const float* __restrict__ hyp_b, float* __restrict__ ne) {
    int idx = blockIdx.x * blockDim.x + threadIdx.x;
    if (idx >= B_ * N_ * EMB_) return;
    int c  = idx % EMB_;
    int bn = idx / EMB_;
    float acc = hyp_b[c];
    for (int o = 0; o < HID_; ++o)
        acc += h[(size_t)bn * HID_ + o] * hyp_W[o * EMB_ + c];
    ne[idx] = acc;
}

// --------------------------------------- encoder support: softmax(relu(EE^T))
__global__ void __launch_bounds__(256)
support_en_kernel(const float* __restrict__ emb, bf16_t* __restrict__ S) {
    __shared__ float row[N_];
    __shared__ float red[256];
    int n = blockIdx.x, tid = threadIdx.x;
    float e[EMB_];
#pragma unroll
    for (int c = 0; c < EMB_; ++c) e[c] = emb[n * EMB_ + c];
    for (int m = tid; m < N_; m += 256) {
        float d = 0.0f;
#pragma unroll
        for (int c = 0; c < EMB_; ++c) d += e[c] * emb[m * EMB_ + c];
        row[m] = fmaxf(d, 0.0f);
    }
    __syncthreads();
    float mx = -1e30f;
    for (int m = tid; m < N_; m += 256) mx = fmaxf(mx, row[m]);
    red[tid] = mx; __syncthreads();
    for (int s = 128; s > 0; s >>= 1) { if (tid < s) red[tid] = fmaxf(red[tid], red[tid + s]); __syncthreads(); }
    mx = red[0]; __syncthreads();
    float sum = 0.0f;
    for (int m = tid; m < N_; m += 256) { float ex = expf(row[m] - mx); row[m] = ex; sum += ex; }
    red[tid] = sum; __syncthreads();
    for (int s = 128; s > 0; s >>= 1) { if (tid < s) red[tid] += red[tid + s]; __syncthreads(); }
    float inv = 1.0f / red[0];
    for (int m = tid; m < N_; m += 256) S[(size_t)n * N_ + m] = (bf16_t)(row[m] * inv);
}

// ------------------------------- decoder supports: per-batch softmax(relu(.))
__global__ void __launch_bounds__(256)
support_de_kernel(const float* __restrict__ ne, bf16_t* __restrict__ S) {
    __shared__ float row[N_];
    __shared__ float red[256];
    int bn = blockIdx.x, tid = threadIdx.x;
    int b = bn / N_, n = bn % N_;
    const float* nb = ne + (size_t)b * N_ * EMB_;
    float e[EMB_];
#pragma unroll
    for (int c = 0; c < EMB_; ++c) e[c] = nb[n * EMB_ + c];
    for (int m = tid; m < N_; m += 256) {
        float d = 0.0f;
#pragma unroll
        for (int c = 0; c < EMB_; ++c) d += e[c] * nb[m * EMB_ + c];
        row[m] = fmaxf(d, 0.0f);
    }
    __syncthreads();
    float mx = -1e30f;
    for (int m = tid; m < N_; m += 256) mx = fmaxf(mx, row[m]);
    red[tid] = mx; __syncthreads();
    for (int s = 128; s > 0; s >>= 1) { if (tid < s) red[tid] = fmaxf(red[tid], red[tid + s]); __syncthreads(); }
    mx = red[0]; __syncthreads();
    float sum = 0.0f;
    for (int m = tid; m < N_; m += 256) { float ex = expf(row[m] - mx); row[m] = ex; sum += ex; }
    red[tid] = sum; __syncthreads();
    for (int s = 128; s > 0; s >>= 1) { if (tid < s) red[tid] += red[tid + s]; __syncthreads(); }
    float inv = 1.0f / red[0];
    for (int m = tid; m < N_; m += 256)
        S[(size_t)b * N_ * N_ + (size_t)n * N_ + m] = (bf16_t)(row[m] * inv);
}

// ----------------------------------------------------- output projection
__global__ void proj_kernel(const float* __restrict__ h, const float* __restrict__ proj_W,
                            const float* __restrict__ proj_b,
                            float* __restrict__ go, float* __restrict__ out, int t) {
    int idx = blockIdx.x * blockDim.x + threadIdx.x;
    if (idx >= B_ * N_) return;
    int b = idx / N_, n = idx % N_;
    float acc = proj_b[0];
    for (int o = 0; o < HID_; ++o) acc += h[(size_t)idx * HID_ + o] * proj_W[o];
    go[idx] = acc;
    out[((size_t)b * HOR_ + t) * N_ + n] = acc;
}

// =========================================================================
extern "C" void kernel_launch(void* const* d_in, const int* in_sizes, int n_in,
                              void* d_out, int out_size, void* d_ws, size_t ws_size,
                              hipStream_t stream) {
    const float* x      = (const float*)d_in[0];
    // d_in[1] = x_cov : unused by the reference
    const float* y_cov  = (const float*)d_in[2];
    const float* emb    = (const float*)d_in[3];
    const float* enc_Wg = (const float*)d_in[4];
    const float* enc_bg = (const float*)d_in[5];
    const float* enc_Wu = (const float*)d_in[6];
    const float* enc_bu = (const float*)d_in[7];
    const float* hyp_W  = (const float*)d_in[8];
    const float* hyp_b  = (const float*)d_in[9];
    const float* dec_Wg = (const float*)d_in[10];
    const float* dec_bg = (const float*)d_in[11];
    const float* dec_Wu = (const float*)d_in[12];
    const float* dec_bu = (const float*)d_in[13];
    const float* proj_W = (const float*)d_in[14];
    const float* proj_b = (const float*)d_in[15];
    float* out = (float*)d_out;

    char* ws = (char*)d_ws;
    size_t off = 0;
    auto alloc = [&](size_t bytes) -> char* {
        char* p = ws + off;
        off += (bytes + 255) & ~(size_t)255;
        return p;
    };
    bf16_t* S_en = (bf16_t*)alloc((size_t)N_ * N_ * 2);
    bf16_t* S_de = (bf16_t*)alloc((size_t)B_ * N_ * N_ * 2);
    float*  Xf   = (float*) alloc((size_t)N_ * LDP_ * 4);
    bf16_t* XT   = (bf16_t*)alloc((size_t)LDP_ * N_ * 2);
    float*  T1f  = (float*) alloc((size_t)N_ * LDP_ * 4);
    bf16_t* T1T  = (bf16_t*)alloc((size_t)LDP_ * N_ * 2);
    float*  T2f  = (float*) alloc((size_t)N_ * LDP_ * 4);
    float*  zr   = (float*) alloc((size_t)B_ * N_ * GATE_ * 4);
    float*  h    = (float*) alloc((size_t)B_ * N_ * HID_ * 4);
    float*  ne   = (float*) alloc((size_t)B_ * N_ * EMB_ * 4);
    float*  go   = (float*) alloc((size_t)B_ * N_ * 4);
    if (off > ws_size) return;  // insufficient scratch

    const dim3 blk32(32), blk256(256);
    const dim3 gEnc(LDP_ / (16 * NT_), N_ / 16, 1);   // 32 x 64 x 1
    const dim3 gDec(1, N_ / 16, B_);                  //  1 x 64 x 32
    const size_t sDeStride = (size_t)N_ * N_;
    const int packN  = (N_ * LDP_ + 255) / 256;
    const int gatesN = (B_ * N_ * GATE_ + 255) / 256;
    const int candN  = (B_ * N_ * HID_ + 255) / 256;
    const int bnN    = (B_ * N_ + 255) / 256;

    support_en_kernel<<<N_, blk256, 0, stream>>>(emb, S_en);
    zero_kernel<<<candN, blk256, 0, stream>>>(h, B_ * N_ * HID_);

    // -------- encoder: 12 GRU steps with shared support --------
    for (int t = 0; t < T_; ++t) {
        pack_kernel<<<packN, blk256, 0, stream>>>(0, t, x, nullptr, nullptr, h, nullptr, Xf, XT);
        sgemm_wmma<<<gEnc, blk32, 0, stream>>>(S_en, 0, XT,  T1f, T1T, 0);
        sgemm_wmma<<<gEnc, blk32, 0, stream>>>(S_en, 0, T1T, T2f, nullptr, 0);
        gates_kernel<<<gatesN, blk256, 0, stream>>>(Xf, T1f, T2f, enc_Wg, enc_bg, zr);
        pack_kernel<<<packN, blk256, 0, stream>>>(0, t, x, nullptr, nullptr, h, zr, Xf, XT);
        sgemm_wmma<<<gEnc, blk32, 0, stream>>>(S_en, 0, XT,  T1f, T1T, 0);
        sgemm_wmma<<<gEnc, blk32, 0, stream>>>(S_en, 0, T1T, T2f, nullptr, 0);
        cand_kernel<<<candN, blk256, 0, stream>>>(Xf, T1f, T2f, enc_Wu, enc_bu, zr, h);
    }

    // -------- hyper-graph: per-batch decoder supports --------
    ne_kernel<<<(B_ * N_ * EMB_ + 255) / 256, blk256, 0, stream>>>(h, hyp_W, hyp_b, ne);
    support_de_kernel<<<B_ * N_, blk256, 0, stream>>>(ne, S_de);
    zero_kernel<<<bnN, blk256, 0, stream>>>(go, B_ * N_);

    // -------- decoder: 12 GRU steps with per-batch supports --------
    for (int t = 0; t < HOR_; ++t) {
        pack_kernel<<<packN, blk256, 0, stream>>>(1, t, nullptr, y_cov, go, h, nullptr, Xf, XT);
        sgemm_wmma<<<gDec, blk32, 0, stream>>>(S_de, sDeStride, XT,  T1f, T1T, CINP_);
        sgemm_wmma<<<gDec, blk32, 0, stream>>>(S_de, sDeStride, T1T, T2f, nullptr, CINP_);
        gates_kernel<<<gatesN, blk256, 0, stream>>>(Xf, T1f, T2f, dec_Wg, dec_bg, zr);
        pack_kernel<<<packN, blk256, 0, stream>>>(1, t, nullptr, y_cov, go, h, zr, Xf, XT);
        sgemm_wmma<<<gDec, blk32, 0, stream>>>(S_de, sDeStride, XT,  T1f, T1T, CINP_);
        sgemm_wmma<<<gDec, blk32, 0, stream>>>(S_de, sDeStride, T1T, T2f, nullptr, CINP_);
        cand_kernel<<<candN, blk256, 0, stream>>>(Xf, T1f, T2f, dec_Wu, dec_bu, zr, h);
        proj_kernel<<<bnN, blk256, 0, stream>>>(h, proj_W, proj_b, go, out, t);
    }
}
